// DynamicInfoLattices_16569983828643
// MI455X (gfx1250) — matrix operand order repaired and checked by reading, not compiled, not run on hardware
//
#include <hip/hip_runtime.h>
#include <cmath>

#define BB   16
#define LL   1024
#define CC   128
#define HHD  512
#define MROW (BB*LL)       /* 16384 */
#define NPIX (LL*CC)       /* 131072 */
#define NTOT (BB*LL*CC)    /* 2097152 */

typedef _Float16 v16h __attribute__((ext_vector_type(16)));
typedef _Float16 v8h  __attribute__((ext_vector_type(8)));
typedef float    v8f  __attribute__((ext_vector_type(8)));

// ---------------------------------------------------------------- helpers
__device__ __forceinline__ float gelu_tanh(float x) {
    // jax.nn.gelu approximate=True
    return 0.5f * x * (1.0f + tanhf(0.7978845608028654f * (x + 0.044715f * x * x * x)));
}

// ------------------------------------------------- weight transpose + f16
// src[R][Ccols] row-major f32  ->  dst[Ccols][R] row-major f16
__global__ __launch_bounds__(256) void transpose_to_half_kernel(
    const float* __restrict__ src, _Float16* __restrict__ dst, int R, int Ccols) {
    int i = blockIdx.x * 256 + threadIdx.x;
    if (i >= R * Ccols) return;
    int r = i / Ccols, c = i - r * Ccols;
    dst[c * R + r] = (_Float16)src[i];
}

// ------------------------------------------------------------- GEMM 1
// h = gelu(z[16384,128] @ W1 + b1 + tf*tw)  -> hh[16384,512] (f16)
// W1t is [N=512][K=128] f16 (pre-transposed). A converted f32->f16 inline.
// Block: 8 waves; each wave computes a 32x64 tile (2 m-tiles x 4 n-blocks).
// Weight strip (64 cols x 128 K = 16 KB) staged once in LDS per block.
__global__ __launch_bounds__(256) void gemm1_kernel(
    const float* __restrict__ z, const _Float16* __restrict__ w1t,
    const float* __restrict__ b1, const float* __restrict__ tw,
    float tfeat, _Float16* __restrict__ hh) {
    const int K = 128, N = 512;
    __shared__ __align__(32) _Float16 Bs[64 * 128];   // 16 KB

    int nstrip = blockIdx.x & 7;           // 8 strips of 64 cols
    int blockM = (blockIdx.x >> 3) * 256;  // 64 m-groups of 256 rows
    int n0g    = nstrip * 64;

    // stage weight strip: rows n0g..n0g+63 of w1t are contiguous 16 KB
    {
        const v8h* src = (const v8h*)(w1t + (size_t)n0g * K);
        v8h* dst = (v8h*)Bs;
        #pragma unroll
        for (int idx = threadIdx.x; idx < (64 * 128 / 8); idx += 256)
            dst[idx] = src[idx];
    }
    __syncthreads();

    int wave = threadIdx.x >> 5;
    int lane = threadIdx.x & 31;
    int lrow = lane & 15, lhi = lane >> 4;
    int mbase = blockM + wave * 32;

    v8f acc[2][4];
    #pragma unroll
    for (int t = 0; t < 2; ++t)
        #pragma unroll
        for (int j = 0; j < 4; ++j)
            #pragma unroll
            for (int e = 0; e < 8; ++e) acc[t][j][e] = 0.0f;

    #pragma unroll
    for (int k0 = 0; k0 < K; k0 += 32) {
        v16h a[2];
        #pragma unroll
        for (int t = 0; t < 2; ++t) {
            const float* ar = z + (size_t)(mbase + t * 16 + lrow) * K + k0 + lhi * 16;
            #pragma unroll
            for (int e = 0; e < 8; ++e) {
                a[t][e]     = (_Float16)ar[e];
                a[t][8 + e] = (_Float16)ar[8 + e];
            }
        }
        #pragma unroll
        for (int j = 0; j < 4; ++j) {
            v16h b = *(const v16h*)(Bs + (j * 16 + lrow) * K + k0 + lhi * 16);
            #pragma unroll
            for (int t = 0; t < 2; ++t)
                acc[t][j] = __builtin_amdgcn_wmma_f32_16x16x32_f16(
                    false, a[t], false, b, (short)0, acc[t][j], false, false);
        }
    }
    #pragma unroll
    for (int j = 0; j < 4; ++j) {
        int n = n0g + j * 16 + lrow;
        float bias = b1[n] + tfeat * tw[n];
        #pragma unroll
        for (int t = 0; t < 2; ++t)
            #pragma unroll
            for (int r = 0; r < 8; ++r) {
                float x = acc[t][j][r] + bias;
                hh[(size_t)(mbase + t * 16 + r + lhi * 8) * N + n] = (_Float16)gelu_tanh(x);
            }
    }
}

// ------------------------------------------------------------- GEMM 2
// eps = hh[16384,512] @ W2 + b2 -> eps[16384,128] f32
// W2t is [N=128][K=512] f16. Weight strip staged in 2 K-slices of 32 KB.
__global__ __launch_bounds__(256) void gemm2_kernel(
    const _Float16* __restrict__ hh, const _Float16* __restrict__ w2t,
    const float* __restrict__ b2, float* __restrict__ eps) {
    const int K = 512, N = 128;
    __shared__ __align__(32) _Float16 Bs[64 * 256];   // 32 KB

    int nstrip = blockIdx.x & 1;           // 2 strips of 64 cols
    int blockM = (blockIdx.x >> 1) * 256;  // 64 m-groups of 256 rows
    int n0g    = nstrip * 64;

    int wave = threadIdx.x >> 5;
    int lane = threadIdx.x & 31;
    int lrow = lane & 15, lhi = lane >> 4;
    int mbase = blockM + wave * 32;

    v8f acc[2][4];
    #pragma unroll
    for (int t = 0; t < 2; ++t)
        #pragma unroll
        for (int j = 0; j < 4; ++j)
            #pragma unroll
            for (int e = 0; e < 8; ++e) acc[t][j][e] = 0.0f;

    for (int p = 0; p < 2; ++p) {
        __syncthreads();
        // stage K-slice [p*256, p*256+256) of rows n0g..n0g+63
        for (int i = threadIdx.x; i < 2048; i += 256) {
            int row = i >> 5, off = i & 31;
            ((v8h*)(Bs + row * 256))[off] =
                ((const v8h*)(w2t + (size_t)(n0g + row) * K + p * 256))[off];
        }
        __syncthreads();

        #pragma unroll 2
        for (int kk = 0; kk < 256; kk += 32) {
            int k0 = p * 256 + kk;
            v16h a[2];
            #pragma unroll
            for (int t = 0; t < 2; ++t) {
                const _Float16* ar = hh + (size_t)(mbase + t * 16 + lrow) * K + k0 + lhi * 16;
                ((v8h*)&a[t])[0] = *(const v8h*)(ar);
                ((v8h*)&a[t])[1] = *(const v8h*)(ar + 8);
            }
            #pragma unroll
            for (int j = 0; j < 4; ++j) {
                v16h b = *(const v16h*)(Bs + (j * 16 + lrow) * 256 + kk + lhi * 16);
                #pragma unroll
                for (int t = 0; t < 2; ++t)
                    acc[t][j] = __builtin_amdgcn_wmma_f32_16x16x32_f16(
                        false, a[t], false, b, (short)0, acc[t][j], false, false);
            }
        }
    }
    #pragma unroll
    for (int j = 0; j < 4; ++j) {
        int n = n0g + j * 16 + lrow;
        float bias = b2[n];
        #pragma unroll
        for (int t = 0; t < 2; ++t)
            #pragma unroll
            for (int r = 0; r < 8; ++r)
                eps[(size_t)(mbase + t * 16 + r + lhi * 8) * N + n] = acc[t][j][r] + bias;
    }
}

// ----------------------------------------------- entropy map (batch mean |eps|)
__global__ __launch_bounds__(256) void entropy_kernel(
    const float* __restrict__ eps, float* __restrict__ ent) {
    int i = blockIdx.x * 256 + threadIdx.x;
    if (i >= NPIX) return;
    float s = 0.0f;
    #pragma unroll
    for (int b = 0; b < BB; ++b) s += fabsf(eps[(size_t)b * NPIX + i]);
    ent[i] = s * (1.0f / 16.0f);
}

// ----------------------------------------------- w x w mean-pool of ent [L,C]
__global__ __launch_bounds__(256) void pool_kernel(
    const float* __restrict__ ent, float* __restrict__ out, int w, int L2, int C2) {
    int i = blockIdx.x * 256 + threadIdx.x;
    if (i >= L2 * C2) return;
    int l2 = i / C2, c2 = i - l2 * C2;
    float s = 0.0f;
    for (int a = 0; a < w; ++a)
        for (int b = 0; b < w; ++b)
            s += ent[(l2 * w + a) * CC + c2 * w + b];
    out[i] = s / (float)(w * w);
}

// ---------------------------------- exact K-th largest (radix select, >=0 floats)
__global__ __launch_bounds__(256) void kth_select_kernel(
    const float* __restrict__ data, int n, int K, float* __restrict__ outT) {
    __shared__ unsigned cnt[256];
    __shared__ unsigned selBin;
    __shared__ int needNext;
    unsigned prefix = 0;
    int need = K;
    for (int pass = 3; pass >= 0; --pass) {
        int sh = pass * 8;
        unsigned himask = (pass == 3) ? 0u : (0xFFFFFFFFu << (sh + 8));
        for (int i = threadIdx.x; i < 256; i += 256) cnt[i] = 0u;
        __syncthreads();
        for (int i = threadIdx.x; i < n; i += 256) {
            unsigned u = __float_as_uint(data[i]);
            if ((u & himask) == prefix)
                atomicAdd(&cnt[(u >> sh) & 255u], 1u);
        }
        __syncthreads();
        if (threadIdx.x == 0) {
            int cum = 0; unsigned b = 0; int nn = need;
            for (int bin = 255; bin >= 0; --bin) {
                int c = (int)cnt[bin];
                if (cum + c >= need) { b = (unsigned)bin; nn = need - cum; break; }
                cum += c;
            }
            selBin = b; needNext = nn;
        }
        __syncthreads();
        prefix |= (selBin << sh);
        need = needNext;
        __syncthreads();
    }
    if (threadIdx.x == 0) *outT = __uint_as_float(prefix);
}

// ------------------------------------- variance of (z-y)*mask, deterministic
__global__ __launch_bounds__(256) void var_partial_kernel(
    const float* __restrict__ z, const float* __restrict__ y,
    const float* __restrict__ m, float* __restrict__ part) {
    __shared__ float ss[256], ss2[256];
    int tid = threadIdx.x;
    float s = 0.0f, s2 = 0.0f;
    for (int i = blockIdx.x * 256 + tid; i < NTOT; i += 256 * 256) {
        float x = (z[i] - y[i]) * m[i];
        s += x; s2 += x * x;
    }
    ss[tid] = s; ss2[tid] = s2; __syncthreads();
    for (int o = 128; o > 0; o >>= 1) {
        if (tid < o) { ss[tid] += ss[tid + o]; ss2[tid] += ss2[tid + o]; }
        __syncthreads();
    }
    if (tid == 0) { part[blockIdx.x] = ss[0]; part[256 + blockIdx.x] = ss2[0]; }
}

__global__ __launch_bounds__(256) void var_final_kernel(
    const float* __restrict__ part, float* __restrict__ S) {
    __shared__ float ss[256], ss2[256];
    int tid = threadIdx.x;
    ss[tid] = part[tid]; ss2[tid] = part[256 + tid]; __syncthreads();
    for (int o = 128; o > 0; o >>= 1) {
        if (tid < o) { ss[tid] += ss[tid + o]; ss2[tid] += ss2[tid + o]; }
        __syncthreads();
    }
    if (tid == 0) {
        const float N = (float)NTOT;
        float mu = ss[0] / N;
        S[4] = ss2[0] / N - mu * mu;
    }
}

// ----------------------------- fused lattice update (finest selected scale wins)
__global__ __launch_bounds__(256) void update_kernel(
    const float* __restrict__ zin, const float* __restrict__ y,
    const float* __restrict__ mask, const float* __restrict__ eps,
    const float* __restrict__ ent, const float* __restrict__ p1,
    const float* __restrict__ p2, const float* __restrict__ p3,
    const float* __restrict__ S, float base_a, float base_b, float h_lam,
    float* __restrict__ zout) {
    int i = blockIdx.x * 256 + threadIdx.x;
    if (i >= NPIX) return;
    int l = i >> 7, c = i & 127;
    float T0 = S[0], T1 = S[1], T2 = S[2], T3 = S[3], var = S[4];

    float entSel = 0.0f; int any = 0;
    float e0 = ent[i];
    if (e0 >= T0) { entSel = e0; any = 1; }
    else {
        float v1 = p1[(l >> 1) * 64 + (c >> 1)];
        if (v1 >= T1) { entSel = v1; any = 1; }
        else {
            float v2 = p2[(l >> 2) * 32 + (c >> 2)];
            if (v2 >= T2) { entSel = v2; any = 1; }
            else {
                float v3 = p3[(l >> 3) * 16 + (c >> 3)];
                if (v3 >= T3) { entSel = v3; any = 1; }
            }
        }
    }
    if (any) {
        float order = 1.0f + (entSel > 0.1f ? 1.0f : 0.0f) + (entSel > 0.5f ? 1.0f : 0.0f);
        float corr  = 1.0f + (order - 1.0f) * 0.5f * h_lam
                    + (order - 1.0f) * (order - 2.0f) * (1.0f / 6.0f) * h_lam * h_lam;
        float gs  = 2.0f * entSel / (entSel + 1.0f);
        float inv = 1.0f / (var + 1e-8f);
        float bbc = base_b * corr;
        for (int b = 0; b < BB; ++b) {
            size_t idx = (size_t)b * NPIX + i;
            float zv = zin[idx];
            float g  = -(zv - y[idx]) * mask[idx] * inv;
            zout[idx] = base_a * zv - bbc * eps[idx] + gs * g;
        }
    } else {
        for (int b = 0; b < BB; ++b) {
            size_t idx = (size_t)b * NPIX + i;
            zout[idx] = zin[idx];
        }
    }
}

// =====================================================================
extern "C" void kernel_launch(void* const* d_in, const int* in_sizes, int n_in,
                              void* d_out, int out_size, void* d_ws, size_t ws_size,
                              hipStream_t stream) {
    (void)in_sizes; (void)n_in; (void)out_size; (void)ws_size;
    const float* y_obs = (const float*)d_in[0];
    const float* mask  = (const float*)d_in[1];
    const float* z_init= (const float*)d_in[2];
    const float* W1    = (const float*)d_in[3];
    const float* b1    = (const float*)d_in[4];
    const float* W2    = (const float*)d_in[5];
    const float* b2    = (const float*)d_in[6];
    const float* tw    = (const float*)d_in[7];

    char* ws = (char*)d_ws;
    const size_t OFF_W1T = 0;                        // 512*128*2   = 131072
    const size_t OFF_W2T = OFF_W1T + 131072;         // 128*512*2   = 131072
    const size_t OFF_HH  = OFF_W2T + 131072;         // 16384*512*2 = 16 MiB
    const size_t OFF_EPS = OFF_HH  + 16777216;       // 16384*128*4 = 8 MiB
    const size_t OFF_ENT = OFF_EPS + 8388608;        // 131072*4
    const size_t OFF_P1  = OFF_ENT + 524288;         // 32768*4
    const size_t OFF_P2  = OFF_P1  + 131072;         // 8192*4
    const size_t OFF_P3  = OFF_P2  + 32768;          // 2048*4
    const size_t OFF_PART= OFF_P3  + 8192;           // 512*4
    const size_t OFF_S   = OFF_PART+ 2048;           // 16*4
    const size_t OFF_ZA  = OFF_S   + 64;             // 2097152*4

    _Float16* W1t = (_Float16*)(ws + OFF_W1T);
    _Float16* W2t = (_Float16*)(ws + OFF_W2T);
    _Float16* hh  = (_Float16*)(ws + OFF_HH);
    float* eps    = (float*)(ws + OFF_EPS);
    float* ent    = (float*)(ws + OFF_ENT);
    float* p1     = (float*)(ws + OFF_P1);
    float* p2     = (float*)(ws + OFF_P2);
    float* p3     = (float*)(ws + OFF_P3);
    float* part   = (float*)(ws + OFF_PART);
    float* S      = (float*)(ws + OFF_S);
    float* zA     = (float*)(ws + OFF_ZA);

    // ---- diffusion schedule (fp32, mirrors jnp) ----
    static float alpha_[1000], sigma_[1000], lam_[1000];
    float ac = 1.0f;
    for (int i = 0; i < 1000; ++i) {
        float beta = 1e-4f + (0.02f - 1e-4f) * (float)i / 999.0f;
        ac *= (1.0f - beta);
        alpha_[i] = sqrtf(ac);
        sigma_[i] = sqrtf(1.0f - ac);
        lam_[i]   = logf(alpha_[i]) - logf(sigma_[i]);
    }

    // weight prep (once per launch)
    transpose_to_half_kernel<<<256, 256, 0, stream>>>(W1, W1t, 128, 512);
    transpose_to_half_kernel<<<256, 256, 0, stream>>>(W2, W2t, 512, 128);

    const float* zcur = z_init;
    for (int s = 0; s < 4; ++s) {
        int k  = 999 - s * 250;
        int kp = k - 250; if (kp < 0) kp = 0;
        float h_lam  = lam_[kp] - lam_[k];
        float base_a = alpha_[kp] / alpha_[k];
        float base_b = sigma_[kp] * (expf(h_lam) - 1.0f);
        float tf     = (float)k / 1000.0f;

        gemm1_kernel<<<512, 256, 0, stream>>>(zcur, W1t, b1, tw, tf, hh);
        gemm2_kernel<<<128, 256, 0, stream>>>(hh, W2t, b2, eps);
        entropy_kernel<<<NPIX / 256, 256, 0, stream>>>(eps, ent);

        pool_kernel<<<128, 256, 0, stream>>>(ent, p1, 2, 512, 64);
        pool_kernel<<<32, 256, 0, stream>>>(ent, p2, 4, 256, 32);
        pool_kernel<<<8, 256, 0, stream>>>(ent, p3, 8, 128, 16);

        kth_select_kernel<<<1, 256, 0, stream>>>(ent, 131072, 26214, S + 0);
        kth_select_kernel<<<1, 256, 0, stream>>>(p1,  32768,  6553, S + 1);
        kth_select_kernel<<<1, 256, 0, stream>>>(p2,   8192,  1638, S + 2);
        kth_select_kernel<<<1, 256, 0, stream>>>(p3,   2048,   409, S + 3);

        var_partial_kernel<<<256, 256, 0, stream>>>(zcur, y_obs, mask, part);
        var_final_kernel<<<1, 256, 0, stream>>>(part, S);

        float* zout = (s == 3) ? (float*)d_out : zA;   // steps 1,2 update in place
        update_kernel<<<NPIX / 256, 256, 0, stream>>>(zcur, y_obs, mask, eps,
                                                      ent, p1, p2, p3, S,
                                                      base_a, base_b, h_lam, zout);
        zcur = zout;
    }
}
